// MultiHeadAttention_36378372997579
// MI455X (gfx1250) — matrix operand order repaired and checked by reading, not compiled, not run on hardware
//
#include <hip/hip_runtime.h>
#include <hip/hip_bf16.h>
#include <math.h>

typedef __attribute__((ext_vector_type(16))) _Float16 v16h;
typedef __attribute__((ext_vector_type(8)))  _Float16 v8h;
typedef __attribute__((ext_vector_type(4)))  _Float16 v4h;
typedef __attribute__((ext_vector_type(8)))  float    v8f;

#define DMODEL 1024
#define NHEADS 16
#define DHEAD  64
#define BATCH  2
#define SEQ    2048
#define MROWS  (BATCH * SEQ)   // 4096

// ---------------------------------------------------------------------------
// WMMA helpers (CDNA5 wave32, V_WMMA_F32_16X16X32_F16)
// ---------------------------------------------------------------------------
__device__ __forceinline__ v8f wmma32(v16h a, v16h b, v8f c) {
  return __builtin_amdgcn_wmma_f32_16x16x32_f16(
      /*neg_a=*/false, a, /*neg_b=*/false, b,
      /*c_mod=*/(short)0, c, /*reuse_a=*/false, /*reuse_b=*/false);
}

// A-matrix 16x32 f16 fragment from row-major [rows, ld] storage.
// Lane l<16 : M=l,   K in {0..15}; lane>=16 : M=l-16, K in {16..31}
__device__ __forceinline__ v16h load_frag_a(const _Float16* base, int row0,
                                            int ld, int k0) {
  int lane = threadIdx.x & 31;
  int m  = lane & 15;
  int kb = (lane >> 4) * 16;
  const _Float16* p = base + (size_t)(row0 + m) * ld + (k0 + kb);
  v8h lo = *(const v8h*)p;          // K = kb .. kb+7
  v8h hi = *(const v8h*)(p + 8);    // K = kb+8 .. kb+15
  v16h a;
#pragma unroll
  for (int i = 0; i < 8; ++i) { a[i] = lo[i]; a[i + 8] = hi[i]; }
  return a;
}

// B-matrix 32x16 f16 fragment from COLUMN-contiguous storage: base[(col)*ld + k].
// Lane l<16 : N=l, K=0..15 ; lane>=16 : N=l-16, K=16..31  -> one 32B load.
__device__ __forceinline__ v16h load_frag_b(const _Float16* base, int col0,
                                            int ld, int k0) {
  int lane = threadIdx.x & 31;
  int c  = lane & 15;
  int kb = (lane >> 4) * 16;
  const _Float16* p = base + (size_t)(col0 + c) * ld + (k0 + kb);
  return *(const v16h*)p;
}

// ---------------------------------------------------------------------------
// f32 -> f16 conversion (elementwise, 4/thread)
// ---------------------------------------------------------------------------
__global__ void mha_cvt_f16_kernel(const float* __restrict__ in,
                                   _Float16* __restrict__ out, int n) {
  int i = (blockIdx.x * blockDim.x + threadIdx.x) * 4;
  if (i + 3 < n) {
    float4 v = *(const float4*)(in + i);
    v4h o; o[0] = (_Float16)v.x; o[1] = (_Float16)v.y;
           o[2] = (_Float16)v.z; o[3] = (_Float16)v.w;
    *(v4h*)(out + i) = o;
  } else {
    for (; i < n; ++i) out[i] = (_Float16)in[i];
  }
}

// ---------------------------------------------------------------------------
// GEMM: C[M,N] = A[M,K] (f16, row-major) @ W[N,K]^T (f16, row-major) + bias.
// Each wave computes a 16x64 tile (4 WMMA N-blocks), K-loop in steps of 32.
// mode 0: store f16 row-major [M,N]
// mode 1: store f16 transposed per head: Vt[b,h,d,s]  (for attention P@V)
// mode 2: store f32 row-major [M,N]  (final output)
// ---------------------------------------------------------------------------
__global__ __launch_bounds__(256)
void mha_gemm_wmma_kernel(const _Float16* __restrict__ A,
                          const _Float16* __restrict__ W,
                          const float* __restrict__ bias,
                          void* __restrict__ outp,
                          int M, int N, int K, int mode) {
  int wave = (int)((blockIdx.x * blockDim.x + threadIdx.x) >> 5);
  int ntn  = N / 64;
  int mt   = wave / ntn;
  int nt   = wave % ntn;
  if (mt * 16 >= M) return;
  int lane = threadIdx.x & 31;
  int hi = lane >> 4, nn = lane & 15;

  v8f acc[4] = {v8f{}, v8f{}, v8f{}, v8f{}};
  for (int k0 = 0; k0 < K; k0 += 32) {
    v16h a = load_frag_a(A, mt * 16, K, k0);
#pragma unroll
    for (int nb = 0; nb < 4; ++nb) {
      v16h b = load_frag_b(W, nt * 64 + nb * 16, K, k0);
      acc[nb] = wmma32(a, b, acc[nb]);
    }
  }
#pragma unroll
  for (int nb = 0; nb < 4; ++nb) {
    int ncol = nt * 64 + nb * 16 + nn;
    float bv = bias ? bias[ncol] : 0.0f;
#pragma unroll
    for (int r = 0; r < 8; ++r) {
      int m = mt * 16 + r + 8 * hi;
      float val = acc[nb][r] + bv;
      if (mode == 2) {
        ((float*)outp)[(size_t)m * N + ncol] = val;
      } else if (mode == 0) {
        ((_Float16*)outp)[(size_t)m * N + ncol] = (_Float16)val;
      } else {  // mode 1: Vt[b, h, d, s]
        int b = m / SEQ, s = m % SEQ;
        int h = ncol / DHEAD, d = ncol % DHEAD;
        ((_Float16*)outp)[(((size_t)(b * NHEADS + h)) * DHEAD + d) * SEQ + s] =
            (_Float16)val;
      }
    }
  }
}

// ---------------------------------------------------------------------------
// Fused attention: one wave owns a 16-row query tile for one (b, h).
// Streams masked scores to HBM once; flash-style online softmax in registers;
// probabilities bounced through a private 1KB LDS tile to form A-fragments.
// ---------------------------------------------------------------------------
__global__ __launch_bounds__(256)
void mha_attn_wmma_kernel(const _Float16* __restrict__ Qh,   // [MROWS, DMODEL]
                          const _Float16* __restrict__ Kh,   // [MROWS, DMODEL]
                          const _Float16* __restrict__ Vt,   // [B, H, DHEAD, SEQ]
                          const unsigned char* __restrict__ mask,  // [SEQ, SEQ]
                          float* __restrict__ scores_out,    // [B, H, SEQ, SEQ]
                          _Float16* __restrict__ ctx_out) {  // [MROWS, DMODEL]
  __shared__ __align__(32) _Float16 plds[8 * 16 * 32];
  int wib  = threadIdx.x >> 5;
  int wave = blockIdx.x * 8 + wib;
  int qt = wave & 127;          // SEQ/16 tiles
  int h  = (wave >> 7) & 15;
  int b  = wave >> 11;
  int lane = threadIdx.x & 31;
  int hi = lane >> 4, nn = lane & 15;
  const int q0 = qt * 16;

  const _Float16* qbase = Qh + (size_t)b * SEQ * DMODEL + h * DHEAD;
  const _Float16* kbase = Kh + (size_t)b * SEQ * DMODEL + h * DHEAD;
  const _Float16* vbase = Vt + (size_t)(b * NHEADS + h) * DHEAD * SEQ;
  float* sout = scores_out + (size_t)(b * NHEADS + h) * SEQ * SEQ;
  _Float16* P = plds + wib * (16 * 32);

  // Q fragments (reused across all key tiles): 16 x 64 = two K=32 A-frags
  v16h qa0 = load_frag_a(qbase, q0, DMODEL, 0);
  v16h qa1 = load_frag_a(qbase, q0, DMODEL, 32);

  float mrow[8], lrow[8];
#pragma unroll
  for (int r = 0; r < 8; ++r) { mrow[r] = -3.0e38f; lrow[r] = 0.0f; }
  v8f acc[4] = {v8f{}, v8f{}, v8f{}, v8f{}};

  for (int sk0 = 0; sk0 < SEQ; sk0 += 32) {
    // ---- scores: two 16x16 tiles, Q(16x64) @ K_tile^T ----
    v8f s[2];
#pragma unroll
    for (int t = 0; t < 2; ++t) {
      v16h kb0 = load_frag_b(kbase, sk0 + t * 16, DMODEL, 0);
      v16h kb1 = load_frag_b(kbase, sk0 + t * 16, DMODEL, 32);
      v8f sc = v8f{};
      sc = wmma32(qa0, kb0, sc);
      sc = wmma32(qa1, kb1, sc);
      s[t] = sc;
    }
    // ---- scale + mask + stream masked scores to HBM ----
#pragma unroll
    for (int t = 0; t < 2; ++t) {
#pragma unroll
      for (int r = 0; r < 8; ++r) {
        int qm = q0 + r + 8 * hi;
        int sk = sk0 + t * 16 + nn;
        float v = s[t][r] * 0.125f;  // 1/sqrt(64)
        if (mask[(size_t)qm * SEQ + sk]) v = -1.0e9f;
        s[t][r] = v;
        sout[(size_t)qm * SEQ + sk] = v;
      }
    }
    // ---- online softmax: row max over 32 keys (16-lane xor reductions) ----
    float nmax[8], alpha[8];
#pragma unroll
    for (int r = 0; r < 8; ++r) {
      float v = fmaxf(s[0][r], s[1][r]);
#pragma unroll
      for (int off = 1; off < 16; off <<= 1)
        v = fmaxf(v, __shfl_xor(v, off, 32));
      nmax[r] = fmaxf(mrow[r], v);
      alpha[r] = __expf(mrow[r] - nmax[r]);
      mrow[r] = nmax[r];
    }
    // ---- exponentiate, write P tile to LDS (C-frag -> row-major) ----
#pragma unroll
    for (int t = 0; t < 2; ++t) {
#pragma unroll
      for (int r = 0; r < 8; ++r) {
        float p = __expf(s[t][r] - mrow[r]);
        s[t][r] = p;
        P[(r + 8 * hi) * 32 + t * 16 + nn] = (_Float16)p;
      }
    }
    // ---- row-sum update ----
#pragma unroll
    for (int r = 0; r < 8; ++r) {
      float v = s[0][r] + s[1][r];
#pragma unroll
      for (int off = 1; off < 16; off <<= 1) v += __shfl_xor(v, off, 32);
      lrow[r] = lrow[r] * alpha[r] + v;
    }
    // ---- rescale context accumulators ----
#pragma unroll
    for (int nb = 0; nb < 4; ++nb)
#pragma unroll
      for (int r = 0; r < 8; ++r) acc[nb][r] *= alpha[r];
    // ---- ctx += P(16x32) @ V(32x64) ----
    v16h pa = load_frag_a(P, 0, 32, 0);
#pragma unroll
    for (int nb = 0; nb < 4; ++nb) {
      v16h vb = load_frag_b(vbase, nb * 16, SEQ, sk0);
      acc[nb] = wmma32(pa, vb, acc[nb]);
    }
  }

  // ---- normalize and emit f16 context, layout [b, s, h, d] = [MROWS, DMODEL]
#pragma unroll
  for (int nb = 0; nb < 4; ++nb) {
#pragma unroll
    for (int r = 0; r < 8; ++r) {
      int qm = q0 + r + 8 * hi;
      float val = acc[nb][r] / lrow[r];
      ctx_out[(size_t)(b * SEQ + qm) * DMODEL + h * DHEAD + nb * 16 + nn] =
          (_Float16)val;
    }
  }
}

// ---------------------------------------------------------------------------
// Host-side launch
// ---------------------------------------------------------------------------
extern "C" void kernel_launch(void* const* d_in, const int* in_sizes, int n_in,
                              void* d_out, int out_size, void* d_ws,
                              size_t ws_size, hipStream_t stream) {
  const float* q_in = (const float*)d_in[0];
  const float* k_in = (const float*)d_in[1];
  const float* v_in = (const float*)d_in[2];
  const unsigned char* mask = (const unsigned char*)d_in[3];
  const float* Wq = (const float*)d_in[4];
  const float* bq = (const float*)d_in[5];
  const float* Wk = (const float*)d_in[6];
  const float* bk = (const float*)d_in[7];
  const float* Wv = (const float*)d_in[8];
  const float* bv = (const float*)d_in[9];
  const float* Wo = (const float*)d_in[10];
  const float* bo = (const float*)d_in[11];

  const size_t MB = 1024 * 1024;
  char* ws = (char*)d_ws;
  _Float16* hQin = (_Float16*)(ws + 0 * MB);    // 8MB each input
  _Float16* hKin = (_Float16*)(ws + 8 * MB);
  _Float16* hVin = (_Float16*)(ws + 16 * MB);
  _Float16* hWq  = (_Float16*)(ws + 24 * MB);   // 2MB each weight
  _Float16* hWk  = (_Float16*)(ws + 26 * MB);
  _Float16* hWv  = (_Float16*)(ws + 28 * MB);
  _Float16* hWo  = (_Float16*)(ws + 30 * MB);
  _Float16* hQ   = (_Float16*)(ws + 32 * MB);   // projected Q f16
  _Float16* hK   = (_Float16*)(ws + 40 * MB);   // projected K f16
  _Float16* hVt  = (_Float16*)(ws + 48 * MB);   // projected V, [B,H,D,S]
  _Float16* hCtx = (_Float16*)(ws + 56 * MB);   // attention context f16

  float* out_f32    = (float*)d_out;                     // [MROWS, DMODEL]
  float* scores_f32 = (float*)d_out + (size_t)MROWS * DMODEL;  // [B,H,S,S]

  const int nInp = MROWS * DMODEL;   // 4,194,304
  const int nW   = DMODEL * DMODEL;  // 1,048,576

  // 1) f32 -> f16 conversions
  mha_cvt_f16_kernel<<<nInp / 1024, 256, 0, stream>>>(q_in, hQin, nInp);
  mha_cvt_f16_kernel<<<nInp / 1024, 256, 0, stream>>>(k_in, hKin, nInp);
  mha_cvt_f16_kernel<<<nInp / 1024, 256, 0, stream>>>(v_in, hVin, nInp);
  mha_cvt_f16_kernel<<<nW / 1024, 256, 0, stream>>>(Wq, hWq, nW);
  mha_cvt_f16_kernel<<<nW / 1024, 256, 0, stream>>>(Wk, hWk, nW);
  mha_cvt_f16_kernel<<<nW / 1024, 256, 0, stream>>>(Wv, hWv, nW);
  mha_cvt_f16_kernel<<<nW / 1024, 256, 0, stream>>>(Wo, hWo, nW);

  // 2) projections: (MROWS/16)*(DMODEL/64) = 4096 waves -> 512 blocks
  const int gemmBlocks = (MROWS / 16) * (DMODEL / 64) / 8;
  mha_gemm_wmma_kernel<<<gemmBlocks, 256, 0, stream>>>(
      hQin, hWq, bq, hQ, MROWS, DMODEL, DMODEL, /*mode=*/0);
  mha_gemm_wmma_kernel<<<gemmBlocks, 256, 0, stream>>>(
      hKin, hWk, bk, hK, MROWS, DMODEL, DMODEL, /*mode=*/0);
  mha_gemm_wmma_kernel<<<gemmBlocks, 256, 0, stream>>>(
      hVin, hWv, bv, hVt, MROWS, DMODEL, DMODEL, /*mode=*/1);

  // 3) fused attention: B*H*(S/16) = 4096 waves -> 512 blocks
  mha_attn_wmma_kernel<<<512, 256, 0, stream>>>(hQ, hK, hVt, mask, scores_f32,
                                                hCtx);

  // 4) output projection (f32 result + bias)
  mha_gemm_wmma_kernel<<<gemmBlocks, 256, 0, stream>>>(
      hCtx, hWo, bo, out_f32, MROWS, DMODEL, DMODEL, /*mode=*/2);
}
